// MultiHeadAttention_12498354831713
// MI455X (gfx1250) — matrix operand order repaired and checked by reading, not compiled, not run on hardware
//
#include <hip/hip_runtime.h>

// ---------------------------------------------------------------------------
// MultiHeadAttention for MI455X (gfx1250, wave32, WMMA f16->f32, TDM staging)
// B=4, S=2048, D=1024, H=16, DK=64
// ---------------------------------------------------------------------------

typedef __attribute__((ext_vector_type(16))) _Float16 v16h;
typedef __attribute__((ext_vector_type(8)))  _Float16 v8h;
typedef __attribute__((ext_vector_type(4)))  _Float16 v4h;
typedef __attribute__((ext_vector_type(8)))  float    v8f;
typedef __attribute__((ext_vector_type(4)))  unsigned int u32x4;
typedef __attribute__((ext_vector_type(8)))  int          i32x8;
typedef __attribute__((ext_vector_type(4)))  int          i32x4;

union F8  { v8f v; float f[8]; };
union H16 { v16h v; v8h h[2]; };

__device__ __forceinline__ v16h make16(v8h lo, v8h hi) {
  H16 u; u.h[0] = lo; u.h[1] = hi; return u.v;
}

#define WMMA_F16(a, b, c) \
  __builtin_amdgcn_wmma_f32_16x16x32_f16(false, (a), false, (b), (short)0, (c), false, false)

#if __has_builtin(__builtin_amdgcn_tensor_load_to_lds) && \
    __has_builtin(__builtin_amdgcn_s_wait_tensorcnt)
#define HAVE_TDM 1
#else
#define HAVE_TDM 0
#endif

static constexpr int Bb = 4, Ss = 2048, Dd = 1024, Hh = 16, DK = 64;
static constexpr int Mtot = Bb * Ss;          // 8192
static constexpr int LDT  = 40;               // GEMM LDS tile row stride (halfs)
static constexpr int LKV  = 72;               // attention LDS row stride (halfs)

// ---------------------------------------------------------------------------
// GEMM: Y[M x 1024] = A[M x 1024] * W^T + bias ; W is [1024(N) x 1024(K)] row-major.
// OUT_MODE 0: f16 output in head-split [B][H][S][DK] layout (for Q/K/V)
// OUT_MODE 1: f32 output, plain [M][N] (final projection -> d_out)
// ---------------------------------------------------------------------------
template <int OUT_MODE, int A_F16>
__global__ __launch_bounds__(256) void mha_proj_gemm(const void* __restrict__ Aptr,
                                                     const float* __restrict__ W,
                                                     const float* __restrict__ bias,
                                                     void* __restrict__ Out) {
  __shared__ _Float16 sA[128 * LDT];
  __shared__ _Float16 sB[128 * LDT];

  const int tid    = threadIdx.x;
  const int bm     = blockIdx.y * 128;
  const int bn     = blockIdx.x * 128;
  const int w      = tid >> 5;
  const int lane   = tid & 31;
  const int hfl    = lane >> 4;
  const int lanelo = lane & 15;
  const int wm     = (w >> 1) * 32;
  const int wn     = (w & 1) * 64;

  F8 acc[2][4];
  for (int mt = 0; mt < 2; ++mt)
    for (int nt = 0; nt < 4; ++nt)
      for (int i = 0; i < 8; ++i) acc[mt][nt].f[i] = 0.0f;

  for (int k0 = 0; k0 < 1024; k0 += 32) {
    __syncthreads();
    if (A_F16) {
      const _Float16* A16 = (const _Float16*)Aptr;
      for (int i = 0; i < 2; ++i) {
        int chunk = tid + i * 256;
        int row = chunk >> 2, c8 = (chunk & 3) * 8;
        *(v8h*)&sA[row * LDT + c8] =
            *(const v8h*)&A16[(size_t)(bm + row) * 1024 + k0 + c8];
      }
    } else {
      const float* A32 = (const float*)Aptr;
      for (int i = 0; i < 4; ++i) {
        int chunk = tid + i * 256;
        int row = chunk >> 3, c4 = (chunk & 7) * 4;
        float4 f = *(const float4*)&A32[(size_t)(bm + row) * 1024 + k0 + c4];
        v4h hv = {(_Float16)f.x, (_Float16)f.y, (_Float16)f.z, (_Float16)f.w};
        *(v4h*)&sA[row * LDT + c4] = hv;
      }
    }
    for (int i = 0; i < 4; ++i) {
      int chunk = tid + i * 256;
      int row = chunk >> 3, c4 = (chunk & 7) * 4;
      float4 f = *(const float4*)&W[(size_t)(bn + row) * 1024 + k0 + c4];
      v4h hv = {(_Float16)f.x, (_Float16)f.y, (_Float16)f.z, (_Float16)f.w};
      *(v4h*)&sB[row * LDT + c4] = hv;
    }
    __syncthreads();

    v16h af[2], bf[4];
    for (int mt = 0; mt < 2; ++mt) {
      int r = wm + mt * 16 + lanelo;
      v8h lo = *(const v8h*)&sA[r * LDT + hfl * 8];
      v8h hi = *(const v8h*)&sA[r * LDT + 16 + hfl * 8];
      af[mt] = make16(lo, hi);
    }
    for (int nt = 0; nt < 4; ++nt) {
      int r = wn + nt * 16 + lanelo;
      v8h lo = *(const v8h*)&sB[r * LDT + hfl * 8];
      v8h hi = *(const v8h*)&sB[r * LDT + 16 + hfl * 8];
      bf[nt] = make16(lo, hi);
    }
    for (int mt = 0; mt < 2; ++mt)
      for (int nt = 0; nt < 4; ++nt)
        acc[mt][nt].v = WMMA_F16(af[mt], bf[nt], acc[mt][nt].v);
  }

  for (int mt = 0; mt < 2; ++mt)
    for (int nt = 0; nt < 4; ++nt) {
      int ng = bn + wn + nt * 16 + lanelo;
      float bv = bias[ng];
      for (int v = 0; v < 8; ++v) {
        int mg = bm + wm + mt * 16 + v + 8 * hfl;
        float val = acc[mt][nt].f[v] + bv;
        if (OUT_MODE == 0) {
          int b = mg >> 11, s = mg & (Ss - 1);
          int h = ng >> 6, dk = ng & (DK - 1);
          ((_Float16*)Out)[(((size_t)b * Hh + h) * Ss + s) * DK + dk] = (_Float16)val;
        } else {
          ((float*)Out)[(size_t)mg * 1024 + ng] = val;
        }
      }
    }
}

// ---------------------------------------------------------------------------
// Flash attention: one block = 128 query rows of one (b,h); 8 waves x 16 rows.
// K tile staged by the Tensor Data Mover (TDM) with hardware LDS padding;
// V tile staged/transposed through VGPRs; all matmuls via v_wmma.
// ---------------------------------------------------------------------------
__global__ __launch_bounds__(256) void mha_flash_attn(const _Float16* __restrict__ Qh,
                                                      const _Float16* __restrict__ Kh,
                                                      const _Float16* __restrict__ Vh,
                                                      const int* __restrict__ mask,
                                                      _Float16* __restrict__ Ctx) {
  __shared__ _Float16 sK[64 * LKV];           // [kv][dk], stride 72 halfs (TDM pad)
  __shared__ _Float16 sV[64 * LKV];           // transposed: [dk][kv]
  __shared__ _Float16 sP[8 * 16 * LKV];       // per-wave P tile [16][64]

  const int tid    = threadIdx.x;
  const int w      = tid >> 5;
  const int lane   = tid & 31;
  const int hfl    = lane >> 4;
  const int lanelo = lane & 15;

  const int qb = blockIdx.x & 15;
  const int bh = blockIdx.x >> 4;
  const int b  = bh >> 4, h = bh & 15;

  const _Float16* Qp = Qh + (size_t)bh * Ss * DK;
  const _Float16* Kp = Kh + (size_t)bh * Ss * DK;
  const _Float16* Vp = Vh + (size_t)bh * Ss * DK;
  const int*      mp = mask + b * Ss;

  // ---- Q fragments straight from global (A layout, 16B b128 loads) ----
  v16h qf[2];
  {
    int qrow = qb * 128 + w * 16 + lanelo;
    for (int kk = 0; kk < 2; ++kk) {
      const _Float16* base = &Qp[(size_t)qrow * DK + kk * 32];
      v8h lo = *(const v8h*)&base[hfl * 8];
      v8h hi = *(const v8h*)&base[16 + hfl * 8];
      qf[kk] = make16(lo, hi);
    }
  }

  F8 Oacc[4];
  for (int t = 0; t < 4; ++t)
    for (int i = 0; i < 8; ++i) Oacc[t].f[i] = 0.0f;
  float rmax[8], rsum[8];
  for (int v = 0; v < 8; ++v) { rmax[v] = -1e30f; rsum[v] = 0.0f; }

  const int wbase = w * 16 * LKV;

  for (int kb = 0; kb < Ss; kb += 64) {
    __syncthreads();

#if HAVE_TDM
    // ---- K tile via Tensor Data Mover: 64x64 f16 tile, HW-padded to stride 72 ----
    if (w == 0) {
      unsigned lds_addr = (unsigned)(uintptr_t)(&sK[0]);   // flat low 32b == LDS addr
      unsigned long long ga = (unsigned long long)(uintptr_t)(&Kp[(size_t)kb * DK]);
      // D# group0: count=1 | lds_addr | global_addr[56:0] | type=2 (bits 127:126)
      u32x4 g0 = { 1u, lds_addr,
                   (unsigned)(ga & 0xFFFFFFFFu),
                   (unsigned)(((ga >> 32) & 0x01FFFFFFu) | 0x80000000u) };
      // D# group1:
      //  w0: wg_mask=0 | data_size=1(2B)<<16 | pad_enable<<20 | pad_interval=4(32dw)<<22
      //      | pad_amount=3(4dw)<<25   -> pad 16B after every 128B row
      //  w1: tensor_dim0[15:0]=64 <<16 ; w2: tensor_dim1[15:0]=64 <<16
      //  w3: tile_dim0=64 <<16        ; w4: tile_dim1=64
      //  w5: tensor_dim0_stride=64
      i32x8 g1 = { (int)0x07110000, (int)0x00400000, (int)0x00400000,
                   (int)0x00400000, 0x40, 64, 0, 0 };
      i32x4 gz4 = { 0, 0, 0, 0 };
      i32x8 gz8 = { 0, 0, 0, 0, 0, 0, 0, 0 };
      __builtin_amdgcn_tensor_load_to_lds(g0, g1, gz4, gz4, gz8, 0);
    }
#else
    for (int i = 0; i < 2; ++i) {
      int chunk = tid + i * 256;
      int kv = chunk >> 3, c8 = (chunk & 7) * 8;
      *(v8h*)&sK[kv * LKV + c8] = *(const v8h*)&Kp[(size_t)(kb + kv) * DK + c8];
    }
#endif

    // ---- V tile staged + transposed -> [dk][kv] ----
    for (int i = 0; i < 2; ++i) {
      int chunk = tid + i * 256;
      int kv = chunk >> 3, c8 = (chunk & 7) * 8;
      v8h tmp = *(const v8h*)&Vp[(size_t)(kb + kv) * DK + c8];
      for (int j = 0; j < 8; ++j) sV[(c8 + j) * LKV + kv] = tmp[j];
    }

#if HAVE_TDM
    if (w == 0) __builtin_amdgcn_s_wait_tensorcnt(0);      // s_wait_tensorcnt 0
#endif
    __syncthreads();

    // prefetch next KV tile into cache while we compute (global_prefetch_b8)
    if (kb + 64 < Ss) {
      __builtin_prefetch(&Kp[(size_t)(kb + 64) * DK + tid * 16], 0, 1);
      __builtin_prefetch(&Vp[(size_t)(kb + 64) * DK + tid * 16], 0, 1);
    }

    // ---- scores S = (Q K^T) * 1/sqrt(DK), with mask ----
    F8 sc[4];
    for (int nt = 0; nt < 4; ++nt) {
      v8f c = {0.f, 0.f, 0.f, 0.f, 0.f, 0.f, 0.f, 0.f};
      for (int kk = 0; kk < 2; ++kk) {
        int r = nt * 16 + lanelo;
        v8h lo = *(const v8h*)&sK[r * LKV + kk * 32 + hfl * 8];
        v8h hi = *(const v8h*)&sK[r * LKV + kk * 32 + 16 + hfl * 8];
        c = WMMA_F16(qf[kk], make16(lo, hi), c);
      }
      int mv = mp[kb + nt * 16 + lanelo];
      F8 tmp; tmp.v = c;
      for (int v = 0; v < 8; ++v)
        sc[nt].f[v] = mv ? tmp.f[v] * 0.125f : -1e30f;
    }

    // ---- online softmax (row = v + 8*hfl; 16 cols across lanes of a half) ----
    float corr[8];
    for (int v = 0; v < 8; ++v) {
      float lm = sc[0].f[v];
      for (int nt = 1; nt < 4; ++nt) lm = fmaxf(lm, sc[nt].f[v]);
      for (int m = 1; m <= 8; m <<= 1) lm = fmaxf(lm, __shfl_xor(lm, m, 32));
      float mn = fmaxf(rmax[v], lm);
      corr[v] = __expf(rmax[v] - mn);
      rmax[v] = mn;
      float ps = 0.0f;
      for (int nt = 0; nt < 4; ++nt) {
        float p = __expf(sc[nt].f[v] - mn);
        sc[nt].f[v] = p;
        ps += p;
      }
      for (int m = 1; m <= 8; m <<= 1) ps += __shfl_xor(ps, m, 32);
      rsum[v] = rsum[v] * corr[v] + ps;
    }
    for (int t = 0; t < 4; ++t)
      for (int v = 0; v < 8; ++v) Oacc[t].f[v] *= corr[v];

    // ---- P: C-layout -> A-layout via per-wave LDS (in-order DS within a wave) ----
    for (int nt = 0; nt < 4; ++nt)
      for (int v = 0; v < 8; ++v)
        sP[wbase + (v + 8 * hfl) * LKV + nt * 16 + lanelo] = (_Float16)sc[nt].f[v];
    asm volatile("" ::: "memory");

    // ---- O += P * V ----
    for (int kk = 0; kk < 2; ++kk) {
      v8h lo = *(const v8h*)&sP[wbase + lanelo * LKV + kk * 32 + hfl * 8];
      v8h hi = *(const v8h*)&sP[wbase + lanelo * LKV + kk * 32 + 16 + hfl * 8];
      v16h pf = make16(lo, hi);
      for (int t = 0; t < 4; ++t) {
        v8h vlo = *(const v8h*)&sV[(t * 16 + lanelo) * LKV + kk * 32 + hfl * 8];
        v8h vhi = *(const v8h*)&sV[(t * 16 + lanelo) * LKV + kk * 32 + 16 + hfl * 8];
        Oacc[t].v = WMMA_F16(pf, make16(vlo, vhi), Oacc[t].v);
      }
    }
  }

  // ---- normalize + write merged-head context [B][S][D] as f16 ----
  for (int t = 0; t < 4; ++t)
    for (int v = 0; v < 8; ++v) {
      float val = Oacc[t].f[v] / rsum[v];
      int sg = qb * 128 + w * 16 + v + 8 * hfl;
      Ctx[((size_t)b * Ss + sg) * Dd + h * DK + t * 16 + lanelo] = (_Float16)val;
    }
}

// ---------------------------------------------------------------------------
extern "C" void kernel_launch(void* const* d_in, const int* in_sizes, int n_in,
                              void* d_out, int out_size, void* d_ws, size_t ws_size,
                              hipStream_t stream) {
  const float* q    = (const float*)d_in[0];
  const float* k    = (const float*)d_in[1];
  const float* v    = (const float*)d_in[2];
  const int*   mask = (const int*)  d_in[3];
  const float* Wq   = (const float*)d_in[4];
  const float* bq   = (const float*)d_in[5];
  const float* Wk   = (const float*)d_in[6];
  const float* bk   = (const float*)d_in[7];
  const float* Wv   = (const float*)d_in[8];
  const float* bv   = (const float*)d_in[9];
  const float* Wo   = (const float*)d_in[10];
  const float* bo   = (const float*)d_in[11];
  float* out = (float*)d_out;

  const size_t elems = (size_t)Bb * Hh * Ss * DK;   // 8,388,608 (16 MB f16 each)
  _Float16* Qh  = (_Float16*)d_ws;
  _Float16* Kh  = Qh + elems;
  _Float16* Vh  = Kh + elems;
  _Float16* Ctx = Vh + elems;
  (void)in_sizes; (void)n_in; (void)out_size; (void)ws_size;

  dim3 gg(1024 / 128, Mtot / 128);   // (8, 64)
  dim3 bb(256);

  mha_proj_gemm<0, 0><<<gg, bb, 0, stream>>>((const void*)q, Wq, bq, (void*)Qh);
  mha_proj_gemm<0, 0><<<gg, bb, 0, stream>>>((const void*)k, Wk, bk, (void*)Kh);
  mha_proj_gemm<0, 0><<<gg, bb, 0, stream>>>((const void*)v, Wv, bv, (void*)Vh);

  mha_flash_attn<<<dim3(Bb * Hh * (Ss / 128)), bb, 0, stream>>>(Qh, Kh, Vh, mask, Ctx);

  mha_proj_gemm<1, 1><<<gg, bb, 0, stream>>>((const void*)Ctx, Wo, bo, (void*)out);
}
